// MultiHeadAttention_50551765074006
// MI455X (gfx1250) — matrix operand order repaired and checked by reading, not compiled, not run on hardware
//
#include <hip/hip_runtime.h>
#include <hip/hip_bf16.h>

// ---------------------------------------------------------------------------
// MHA for MI455X (gfx1250, wave32, WMMA + TDM).
// Pipeline:
//   1) gemm128<false,0> x2 : Q/K = f16(x@W + b) into [B,H,S,HD] ws (Q * 1/8)
//      gemm128<false,2>    : V   = f16(x@W + b) into [B,H,HD,S] ws (transposed)
//   2) attn_kernel         : flash attention; V tiles staged to LDS via
//                            tensor_load_to_lds (TDM); ctx f16 [B,S,D] in ws
//   3) gemm128<true,1>     : out = f32(ctx@Wo + bo)
// Workspace: 4 x 8MB f16 buffers = 32 MB.
// ---------------------------------------------------------------------------

typedef __attribute__((ext_vector_type(16))) _Float16     v16h;
typedef __attribute__((ext_vector_type(8)))  _Float16     v8h;
typedef __attribute__((ext_vector_type(4)))  _Float16     v4h;
typedef __attribute__((ext_vector_type(8)))  float        v8f;
typedef __attribute__((ext_vector_type(4)))  float        v4f;
typedef __attribute__((ext_vector_type(4)))  unsigned int v4u;
typedef __attribute__((ext_vector_type(8)))  int          v8i;
typedef __attribute__((ext_vector_type(4)))  int          v4i;

#define NHEAD 16
#define DMODEL 1024
#define HDIM 64
#define SEQ 2048
#define NBATCH 2

// Assemble a 16-half WMMA operand from two contiguous 16B runs.
static __device__ __forceinline__ v16h ld_frag16(const _Float16* p0, const _Float16* p1) {
  union { v16h v; v8h h[2]; } u;
  u.h[0] = *(const v8h*)p0;
  u.h[1] = *(const v8h*)p1;
  return u.v;
}

static __device__ __forceinline__ v8f wmma_f16(v16h a, v16h b, v8f c) {
  // 8 args: (neg_a, A, neg_b, B, c_mod, C, reuse_a, reuse_b)
  return __builtin_amdgcn_wmma_f32_16x16x32_f16(false, a, false, b, (short)0, c, false, false);
}

// ---------------------------------------------------------------------------
// Tiled GEMM: C[M=4096, N=1024] = A[4096,1024] @ W[1024,1024] + bias
// A_HALF: A is f16 (ctx) vs f32 (x).
// EPI==0: f16 [B,H,S,HD] out with scale (Q/K)
// EPI==1: f32 row-major out (final output)
// EPI==2: f16 [B,H,HD,S] transposed out (V), packed 16B stores
// Block 128x128, 256 threads (8 waves, 2x4), wave tile 64x32 (4x2 WMMA accs).
// ---------------------------------------------------------------------------
template<bool A_HALF, int EPI>
__global__ void __launch_bounds__(256)
gemm128(const void* __restrict__ Aptr, const float* __restrict__ W,
        const float* __restrict__ bias, void* __restrict__ outp, float scale)
{
  __shared__ __attribute__((aligned(16))) _Float16 As[128 * 32];  // [m][k]
  __shared__ __attribute__((aligned(16))) _Float16 Bs[128 * 32];  // [n][k] (transposed)

  const int t    = threadIdx.x;
  const int lane = t & 31;
  const int wid  = t >> 5;
  const int m0   = blockIdx.y * 128;
  const int n0   = blockIdx.x * 128;
  const int wm   = (wid >> 2) * 64;   // wave row offset in block
  const int wn   = (wid & 3) * 32;    // wave col offset in block
  const int lcol = lane & 15;
  const int hb8  = (lane >> 4) * 8;   // half-wave row offset (acc)
  const int hb16 = (lane >> 4) * 16;  // half-wave K offset (B frag)

  v8f acc[4][2];
  #pragma unroll
  for (int i = 0; i < 4; ++i)
    #pragma unroll
    for (int j = 0; j < 2; ++j) acc[i][j] = (v8f){};

  for (int kk = 0; kk < DMODEL; kk += 32) {
    // prefetch next k-tile (global_prefetch_b8)
    if (kk + 32 < DMODEL) {
      __builtin_prefetch(W + (size_t)(kk + 32 + (t >> 5)) * DMODEL + n0 + (t & 31) * 4, 0, 1);
      if constexpr (A_HALF)
        __builtin_prefetch((const _Float16*)Aptr + (size_t)(m0 + (t >> 1)) * DMODEL + kk + 32, 0, 1);
      else
        __builtin_prefetch((const float*)Aptr + (size_t)(m0 + (t >> 1)) * DMODEL + kk + 32, 0, 1);
    }
    __syncthreads();
    // --- stage A tile 128x32, convert to f16 ---
    #pragma unroll
    for (int i = 0; i < 4; ++i) {
      const int idx = i * 256 + t;
      const int row = idx >> 3;
      const int c4  = (idx & 7) * 4;
      v4h hv;
      if constexpr (A_HALF) {
        hv = *(const v4h*)((const _Float16*)Aptr + (size_t)(m0 + row) * DMODEL + kk + c4);
      } else {
        const v4f fv = *(const v4f*)((const float*)Aptr + (size_t)(m0 + row) * DMODEL + kk + c4);
        #pragma unroll
        for (int j = 0; j < 4; ++j) hv[j] = (_Float16)fv[j];
      }
      *(v4h*)(As + row * 32 + c4) = hv;
    }
    // --- stage W tile 32x128 transposed into Bs[n][k] ---
    #pragma unroll
    for (int i = 0; i < 4; ++i) {
      const int idx  = i * 256 + t;
      const int krow = idx >> 5;
      const int c4   = (idx & 31) * 4;
      const v4f fv = *(const v4f*)(W + (size_t)(kk + krow) * DMODEL + n0 + c4);
      #pragma unroll
      for (int j = 0; j < 4; ++j) Bs[(c4 + j) * 32 + krow] = (_Float16)fv[j];
    }
    __syncthreads();

    v16h a[4], b[2];
    #pragma unroll
    for (int mi = 0; mi < 4; ++mi) {
      const _Float16* p = As + (wm + mi * 16 + lcol) * 32;
      a[mi] = ld_frag16(p + hb16, p + hb16 + 8);
    }
    #pragma unroll
    for (int ni = 0; ni < 2; ++ni) {
      const _Float16* p = Bs + (wn + ni * 16 + lcol) * 32 + hb16;
      b[ni] = ld_frag16(p, p + 8);
    }
    #pragma unroll
    for (int mi = 0; mi < 4; ++mi)
      #pragma unroll
      for (int ni = 0; ni < 2; ++ni)
        acc[mi][ni] = wmma_f16(a[mi], b[ni], acc[mi][ni]);
  }

  // --- epilogue ---
  #pragma unroll
  for (int mi = 0; mi < 4; ++mi) {
    #pragma unroll
    for (int ni = 0; ni < 2; ++ni) {
      const int col = n0 + wn + ni * 16 + lcol;
      const float bv = bias[col];
      if constexpr (EPI == 2) {
        // V transposed layout [B,H,HD,S]: 8 consecutive rows = consecutive s
        const int row0 = m0 + wm + mi * 16 + hb8;
        const int bI = row0 >> 11, srow = row0 & (SEQ - 1);
        const int hh = col >> 6,  hd = col & (HDIM - 1);
        v8h pk;
        #pragma unroll
        for (int r = 0; r < 8; ++r) pk[r] = (_Float16)(acc[mi][ni][r] + bv);
        *(v8h*)((_Float16*)outp + ((size_t)(bI * NHEAD + hh) * HDIM + hd) * SEQ + srow) = pk;
      } else {
        #pragma unroll
        for (int r = 0; r < 8; ++r) {
          const int row = m0 + wm + mi * 16 + hb8 + r;
          const float val = acc[mi][ni][r] + bv;
          if constexpr (EPI == 0) {
            const int bI = row >> 11, s = row & (SEQ - 1);
            const int hh = col >> 6,  hd = col & (HDIM - 1);
            ((_Float16*)outp)[(((size_t)(bI * NHEAD + hh) * SEQ + s) * HDIM) + hd] =
                (_Float16)(val * scale);
          } else {
            ((float*)outp)[(size_t)row * DMODEL + col] = val;
          }
        }
      }
    }
  }
}

// ---------------------------------------------------------------------------
// Flash attention. Grid: (B*H, S/128). 256 threads = 8 waves, 16 queries/wave.
// Computes S^T = K @ Q^T (C layout: lane&15 = query), then O^T = V^T @ P^T.
// Q (pre-scaled 1/8), K are f16 [B,H,S,HD]; Vt is f16 [B,H,HD,S]; ctx [B,S,D].
// V tiles (64 hd x 32 keys) staged to LDS by the Tensor Data Mover.
// ---------------------------------------------------------------------------
__global__ void __launch_bounds__(256)
attn_kernel(const _Float16* __restrict__ Q, const _Float16* __restrict__ K,
            const _Float16* __restrict__ Vt, _Float16* __restrict__ ctx)
{
  __shared__ __attribute__((aligned(16))) _Float16 Vs[HDIM * 32];      // Vs[hd][key]
  __shared__ __attribute__((aligned(16))) _Float16 Ps[8 * 16 * 32];    // per-wave P[q][key]

  const int bh   = blockIdx.x;
  const int b    = bh >> 4;
  const int h    = bh & 15;
  const int t    = threadIdx.x;
  const int lane = t & 31;
  const int wid  = t >> 5;
  const size_t base = (size_t)bh * SEQ * HDIM;
  const _Float16* Qh = Q + base;
  const _Float16* Kh = K + base;
  const _Float16* Vh = Vt + base;   // [HD][S] tile base for this (b,h)

  const int q0     = blockIdx.y * 128 + wid * 16;   // this wave's query base
  const int lcol   = lane & 15;                     // query within tile
  const int hb8    = (lane >> 4) * 8;
  const int hb16   = (lane >> 4) * 16;
  _Float16* Pw = Ps + wid * (16 * 32);

  // TDM descriptor constants for the V tile copy (64 x 32 halves, row-major)
  const unsigned int lds_addr = (unsigned int)(uintptr_t)Vs;  // LDS byte offset (low 32 bits)
  const unsigned long long ga_base = (unsigned long long)(uintptr_t)Vh;

  // Q^T B-fragments (two 32-deep hd chunks), loaded once from global.
  const _Float16* qrow = Qh + (size_t)(q0 + lcol) * HDIM;
  v16h qf[2];
  #pragma unroll
  for (int c = 0; c < 2; ++c)
    qf[c] = ld_frag16(qrow + c * 32 + hb8, qrow + c * 32 + 16 + hb8);

  v8f o[4];                                         // O^T tiles, hd 0..63
  #pragma unroll
  for (int i = 0; i < 4; ++i) o[i] = (v8f){};
  float m = -3.0e38f, l = 0.0f;

  for (int kb = 0; kb < SEQ; kb += 32) {
    if (kb + 32 < SEQ)
      __builtin_prefetch(Kh + (size_t)(kb + 32 + lane) * HDIM, 0, 1);

    // --- stage V tile [64 hd][32 keys] into Vs via Tensor Data Mover ---
    __syncthreads();                                // prior-iteration reads done
    if (wid == 0) {
      const unsigned long long ga = ga_base + (unsigned long long)kb * 2ull;
      const v4u g0 = {
        1u,                                         // count=1, is_restore=0, gather=0
        lds_addr,                                   // lds_addr
        (unsigned int)(ga & 0xffffffffull),         // global_addr[31:0]
        (unsigned int)((ga >> 32) & 0x1ffffffull) | 0x80000000u  // addr[56:32] | type=2
      };
      const v8i g1 = {
        0x00010000,                                 // data_size=2B, mask=0
        (int)(2048u << 16),                         // tensor_dim0[15:0]=2048 (keys)
        (int)(64u << 16),                           // tensor_dim1[15:0]=64 (hd rows)
        (int)(32u << 16),                           // tile_dim0=32
        64,                                         // tile_dim1=64
        2048,                                       // tensor_dim0_stride=2048
        0, 0
      };
      const v4i g2 = {0, 0, 0, 0};
      const v4i g3 = {0, 0, 0, 0};
      const v8i g4 = {0, 0, 0, 0, 0, 0, 0, 0};
      __builtin_amdgcn_tensor_load_to_lds(g0, g1, g2, g3, g4, 0);
      __builtin_amdgcn_s_wait_tensorcnt(0);
    }
    __syncthreads();                                // Vs ready for all waves

    // --- S^T tiles: s0 = keys kb..kb+15, s1 = keys kb+16..kb+31 ---
    v8f s0 = (v8f){}, s1 = (v8f){};
    #pragma unroll
    for (int c = 0; c < 2; ++c) {
      const _Float16* kr0 = Kh + (size_t)(kb + lcol) * HDIM + c * 32;
      const _Float16* kr1 = Kh + (size_t)(kb + 16 + lcol) * HDIM + c * 32;
      const v16h ka0 = ld_frag16(kr0 + hb8, kr0 + 16 + hb8);
      const v16h ka1 = ld_frag16(kr1 + hb8, kr1 + 16 + hb8);
      s0 = wmma_f16(ka0, qf[c], s0);
      s1 = wmma_f16(ka1, qf[c], s1);
    }

    // --- online softmax (per lane = per query; combine half-waves) ---
    float bm = -3.0e38f;
    #pragma unroll
    for (int r = 0; r < 8; ++r) bm = fmaxf(bm, fmaxf(s0[r], s1[r]));
    bm = fmaxf(bm, __shfl_xor(bm, 16, 32));
    const float mn = fmaxf(m, bm);
    const float sc = __expf(m - mn);
    float ps = 0.0f;
    #pragma unroll
    for (int r = 0; r < 8; ++r) {
      const float e0 = __expf(s0[r] - mn);
      const float e1 = __expf(s1[r] - mn);
      ps += e0 + e1;
      Pw[lcol * 32 + hb8 + r]      = (_Float16)e0;
      Pw[lcol * 32 + 16 + hb8 + r] = (_Float16)e1;
    }
    ps += __shfl_xor(ps, 16, 32);
    l = l * sc + ps;
    m = mn;
    #pragma unroll
    for (int i = 0; i < 4; ++i) o[i] *= sc;

    // --- O^T += V^T @ P^T ---
    const _Float16* pb = Pw + lcol * 32 + hb16;
    const v16h pfrag = ld_frag16(pb, pb + 8);
    #pragma unroll
    for (int tt = 0; tt < 4; ++tt) {
      const _Float16* va = Vs + (tt * 16 + lcol) * 32;
      const v16h vfrag = ld_frag16(va + hb16, va + hb16 + 8);
      o[tt] = wmma_f16(vfrag, pfrag, o[tt]);
    }
  }

  // --- write ctx[b][s][h*64+hd] ---
  const float inv = 1.0f / l;
  const int s = q0 + lcol;
  #pragma unroll
  for (int tt = 0; tt < 4; ++tt)
    #pragma unroll
    for (int r = 0; r < 8; ++r) {
      const int hd = tt * 16 + hb8 + r;
      ctx[((size_t)b * SEQ + s) * DMODEL + h * HDIM + hd] = (_Float16)(o[tt][r] * inv);
    }
}

// ---------------------------------------------------------------------------
extern "C" void kernel_launch(void* const* d_in, const int* in_sizes, int n_in,
                              void* d_out, int out_size, void* d_ws, size_t ws_size,
                              hipStream_t stream) {
  (void)in_sizes; (void)n_in; (void)out_size; (void)ws_size;
  const float* x  = (const float*)d_in[0];
  const float* Wq = (const float*)d_in[1];
  const float* bq = (const float*)d_in[2];
  const float* Wk = (const float*)d_in[3];
  const float* bk = (const float*)d_in[4];
  const float* Wv = (const float*)d_in[5];
  const float* bv = (const float*)d_in[6];
  const float* Wo = (const float*)d_in[7];
  const float* bo = (const float*)d_in[8];

  const size_t elems = (size_t)NBATCH * SEQ * DMODEL;   // 4,194,304
  _Float16* qws = (_Float16*)d_ws;
  _Float16* kws = qws + elems;
  _Float16* vws = kws + elems;
  _Float16* cws = vws + elems;

  const dim3 blk(256);
  const dim3 gproj(DMODEL / 128, (NBATCH * SEQ) / 128);  // (8, 32)
  const float qscale = 0.125f;                           // 1/sqrt(HD)

  gemm128<false, 0><<<gproj, blk, 0, stream>>>(x, Wq, bq, qws, qscale);
  gemm128<false, 0><<<gproj, blk, 0, stream>>>(x, Wk, bk, kws, 1.0f);
  gemm128<false, 2><<<gproj, blk, 0, stream>>>(x, Wv, bv, vws, 1.0f);

  attn_kernel<<<dim3(NBATCH * NHEAD, SEQ / 128), blk, 0, stream>>>(qws, kws, vws, cws);

  gemm128<true, 1><<<gproj, blk, 0, stream>>>(cws, Wo, bo, d_out, 1.0f);
}
